// CharacterFeatures_31275951850196
// MI455X (gfx1250) — matrix operand order repaired and checked by reading, not compiled, not run on hardware
//
#include <hip/hip_runtime.h>
#include <hip/hip_bf16.h>

// Problem constants (match reference)
#define NW    16384        // B*S words
#define TT    20           // chars per word
#define DD    128          // char embedding dim
#define HH    256          // hidden dim
#define G3    768          // 3*H
#define NC    129          // embedding rows (0 = pad)

typedef __attribute__((ext_vector_type(8)))  float  v8f;
typedef __attribute__((ext_vector_type(16))) __bf16 v16bf;

union FragU {
    v16bf v;
    uint4 q[2];
    unsigned short u[16];
};

__device__ __forceinline__ unsigned short f2bf(float f) {
    unsigned int u = __float_as_uint(f);
    u = (u + 0x7fffu + ((u >> 16) & 1u)) >> 16;   // round-to-nearest-even
    return (unsigned short)u;
}
__device__ __forceinline__ float bf2f(unsigned short h) {
    return __uint_as_float(((unsigned int)h) << 16);
}
__device__ __forceinline__ float sigmoidf_(float x) {
    return 1.0f / (1.0f + __expf(-x));
}

// ---------------------------------------------------------------------------
// Prep 1: xg_table[dir][c][j] = emb[c,:] @ Wk[:,j] + b_in[j] (+ b_rec[j] for z,r)
// ---------------------------------------------------------------------------
__global__ void xg_table_kernel(const float* __restrict__ emb,
                                const float* __restrict__ WkF,
                                const float* __restrict__ WkB,
                                const float* __restrict__ bF,
                                const float* __restrict__ bB,
                                float* __restrict__ tab) {
    int idx = blockIdx.x * blockDim.x + threadIdx.x;
    if (idx >= 2 * NC * G3) return;
    int dir = idx / (NC * G3);
    int rem = idx - dir * (NC * G3);
    int c = rem / G3;
    int j = rem - c * G3;
    const float* Wk = dir ? WkB : WkF;
    const float* b  = dir ? bB  : bF;
    float acc = b[j];                         // b_in
    if (j < 2 * HH) acc += b[G3 + j];         // fold b_rec for z and r gates only
    const float* e = emb + c * DD;
    #pragma unroll 8
    for (int d = 0; d < DD; ++d) acc += e[d] * Wk[d * G3 + j];
    tab[idx] = acc;
}

// ---------------------------------------------------------------------------
// Prep 2: repack Wr (fp32, [256 x 768]) into bf16 WMMA-B fragments.
// Fragment (dir, kt, nt, lane) element i = Wr[kt*32 + 16*(i/8) + 8*(lane/16) + i%8][nt*16 + lane%16].
// (B 32x16 bf16 layout: K = 16*(i/8) + 8*(lane/16) + i%8, N = lane%16.)
// ---------------------------------------------------------------------------
__global__ void repack_wr_kernel(const float* __restrict__ WrF,
                                 const float* __restrict__ WrB,
                                 unsigned short* __restrict__ dst) {
    int idx = blockIdx.x * blockDim.x + threadIdx.x;   // 2*8*48*32 = 24576
    if (idx >= 24576) return;
    int lane = idx & 31;
    int t2   = idx >> 5;
    int nt   = t2 % 48;
    int kt   = (t2 / 48) % 8;
    int dir  = t2 / 384;
    const float* Wr  = dir ? WrB : WrF;
    const float* src = Wr + (kt * 32) * G3 + nt * 16 + (lane & 15);
    unsigned short* d = dst + (size_t)idx * 16;
    #pragma unroll
    for (int i = 0; i < 16; ++i) d[i] = f2bf(src[((i & 7) + 8 * (lane >> 4) + 16 * (i >> 3)) * G3]);
}

// ---------------------------------------------------------------------------
// Main GRU kernel.  Grid: (NW/32, 2 directions).  Block: 256 threads = 8 waves.
// Each workgroup owns 32 word-rows; h state is bf16 in LDS [32][256].
// Wave w owns H-columns [32w, 32w+32) across all three gates.
// ---------------------------------------------------------------------------
__global__ __launch_bounds__(256)
void gru_wmma_kernel(const int*   __restrict__ chars,   // [NW][TT]
                     const float* __restrict__ xg_tab,  // [2][NC][768]
                     const unsigned short* __restrict__ wrPk, // [2][8][48][32][16]
                     const float* __restrict__ bF,      // [2][768]
                     const float* __restrict__ bB,
                     float* __restrict__ out) {         // [NW][512]
    const int tid  = threadIdx.x;
    const int wave = tid >> 5;
    const int lane = tid & 31;
    const int dir  = blockIdx.y;
    const int rowBase = blockIdx.x * 32;

    __shared__ unsigned short h_s[32][HH];   // 16 KB bf16 hidden state
    __shared__ int cid_s[32];

    for (int i = tid; i < 32 * HH; i += 256) ((unsigned short*)h_s)[i] = 0;
    __syncthreads();

    const float* xg = xg_tab + (size_t)dir * (NC * G3);
    const unsigned short* wb = wrPk + (size_t)dir * (8 * 48 * 32 * 16);
    const float* brec = (dir ? bB : bF) + G3;       // b_rec row

    const int n  = lane & 15;       // N within tile / M within A row
    const int mh = lane >> 4;       // lane-half selector

    int   colH[2];
    float brh[2];
    #pragma unroll
    for (int u = 0; u < 2; ++u) {
        colH[u] = wave * 32 + u * 16 + n;
        brh[u]  = brec[2 * HH + colH[u]];           // b_rec, h-gate slice
    }

    for (int t = 0; t < TT; ++t) {
        const int tc = dir ? (TT - 1 - t) : t;
        if (tid < 32) cid_s[tid] = chars[(rowBase + tid) * TT + tc];

        v8f acc[2][3][2];
        #pragma unroll
        for (int mt = 0; mt < 2; ++mt)
            #pragma unroll
            for (int g = 0; g < 3; ++g)
                #pragma unroll
                for (int u = 0; u < 2; ++u)
                    acc[mt][g][u] = (v8f)(0.0f);

        // hg = h @ Wr  (bf16 WMMA, fp32 accumulate)
        #pragma unroll 2
        for (int kt = 0; kt < 8; ++kt) {
            FragU a[2];
            const int kbase = kt * 32 + mh * 8;     // A layout: K = 16*(i/8) + 8*mh + i%8
            #pragma unroll
            for (int mt = 0; mt < 2; ++mt) {
                const int row = mt * 16 + n;
                a[mt].q[0] = *(const uint4*)&h_s[row][kbase];
                a[mt].q[1] = *(const uint4*)&h_s[row][kbase + 16];
            }
            #pragma unroll
            for (int g = 0; g < 3; ++g) {
                #pragma unroll
                for (int u = 0; u < 2; ++u) {
                    const int nt = g * 16 + wave * 2 + u;
                    FragU b;
                    const unsigned short* bp = wb + (((size_t)kt * 48 + nt) * 32 + lane) * 16;
                    b.q[0] = *(const uint4*)(bp);
                    b.q[1] = *(const uint4*)(bp + 8);
                    #pragma unroll
                    for (int mt = 0; mt < 2; ++mt)
                        acc[mt][g][u] = __builtin_amdgcn_wmma_f32_16x16x32_bf16(
                            false, a[mt].v, false, b.v, (short)0, acc[mt][g][u],
                            false, false);
                }
            }
        }
        __syncthreads();   // all A-frag reads done; cid_s visible

        // Gate math (each (row,col) owned by exactly one lane)
        #pragma unroll
        for (int mt = 0; mt < 2; ++mt) {
            #pragma unroll
            for (int v = 0; v < 8; ++v) {
                const int row = mt * 16 + v + mh * 8;    // C layout: M = v + 8*(lane/16)
                const int c   = cid_s[row];
                const float* xr = xg + (size_t)c * G3;
                #pragma unroll
                for (int u = 0; u < 2; ++u) {
                    const int col = colH[u];
                    const float hz = acc[mt][0][u][v];
                    const float hr = acc[mt][1][u][v];
                    const float hhv = acc[mt][2][u][v];
                    const float z  = sigmoidf_(xr[col] + hz);
                    const float r  = sigmoidf_(xr[HH + col] + hr);
                    const float hc = tanhf(xr[2 * HH + col] + r * (hhv + brh[u]));
                    const float ho = bf2f(h_s[row][col]);
                    float hn = z * ho + (1.0f - z) * hc;
                    if (c == 0) hn = ho;                 // masked step: pass state through
                    h_s[row][col] = f2bf(hn);
                }
            }
        }
        __syncthreads();   // hn visible before next step's A loads
    }

    // Emit final hidden state: out[:, 0:256]=fwd, out[:, 256:512]=bwd
    #pragma unroll
    for (int mt = 0; mt < 2; ++mt) {
        #pragma unroll
        for (int v = 0; v < 8; ++v) {
            const int row = mt * 16 + v + mh * 8;
            #pragma unroll
            for (int u = 0; u < 2; ++u) {
                const int col = colH[u];
                out[(size_t)(rowBase + row) * (2 * HH) + dir * HH + col] =
                    bf2f(h_s[row][col]);
            }
        }
    }
}

// ---------------------------------------------------------------------------
extern "C" void kernel_launch(void* const* d_in, const int* in_sizes, int n_in,
                              void* d_out, int out_size, void* d_ws, size_t ws_size,
                              hipStream_t stream) {
    const int*   chars = (const int*)  d_in[0];
    const float* emb   = (const float*)d_in[1];
    const float* WkF   = (const float*)d_in[2];
    const float* WrF   = (const float*)d_in[3];
    const float* bF    = (const float*)d_in[4];
    const float* WkB   = (const float*)d_in[5];
    const float* WrB   = (const float*)d_in[6];
    const float* bB    = (const float*)d_in[7];
    float* out = (float*)d_out;

    float* tab = (float*)d_ws;                               // 2*129*768 fp32
    unsigned short* wrPk = (unsigned short*)((char*)d_ws +
                            (size_t)2 * NC * G3 * sizeof(float)); // 2*8*48*512 bf16

    {
        int total = 2 * NC * G3;
        xg_table_kernel<<<(total + 255) / 256, 256, 0, stream>>>(emb, WkF, WkB, bF, bB, tab);
    }
    repack_wr_kernel<<<(24576 + 255) / 256, 256, 0, stream>>>(WrF, WrB, wrPk);

    dim3 grid(NW / 32, 2);
    gru_wmma_kernel<<<grid, 256, 0, stream>>>(chars, tab, wrPk, bF, bB, out);
}
